// MambaBlock_4896262718010
// MI455X (gfx1250) — compile-verified, matrix operand order checked by reading, not run on hardware
//
#include <hip/hip_runtime.h>
#include <hip/hip_bf16.h>

#define BATCH 4
#define SEQ   2048
#define DMOD  1024
#define NL    4
#define NDS   16
#define DCONV 4
#define DI    2048
#define DTR   64
#define XPN   (DTR + 2*NDS)   // 96
#define MROWS (BATCH*SEQ)     // 8192

typedef __attribute__((ext_vector_type(16))) _Float16 v16h;
typedef __attribute__((ext_vector_type(8)))  _Float16 v8h;
typedef __attribute__((ext_vector_type(8)))  float    v8f;

enum { EP_SPLIT = 0, EP_PLAIN = 1, EP_SOFTPLUS = 2, EP_BIAS = 3 };

// ---------------------------------------------------------------------------
// Generic f16 WMMA GEMM:  C[M,N] = A[M,K] * W[N,K]^T  (f32 accumulate)
// Block: 256 thr = 8 waves (4 in M x 2 in N); wave tile 32x64; block 128x128.
// Fragment layout per CDNA5 ISA 7.12.2 (16-bit A 16x32 / B 32x16):
//   lane<16 : row = lane,   K in {k0..k0+7, k0+16..k0+23}
//   lane>=16: row = lane-16,K in {k0+8..k0+15, k0+24..k0+31}
// B rows >= N are clamped to row N-1 (their product lands only in never-stored
// D columns), so the hot loop has no exec-mask manipulation.
// Ping-pong fragment buffers + unroll-by-2 (K is always a multiple of 64):
// loads for buffer B are in flight while the 8 WMMAs of buffer A issue, and
// the buffers are disjoint registers, so no copies / WAR-hazard v_nops.
// ---------------------------------------------------------------------------
template<int EP>
__global__ __launch_bounds__(256)
void gemm_wmma(const _Float16* __restrict__ A, const _Float16* __restrict__ W,
               float* __restrict__ C0, float* __restrict__ C1,
               const float* __restrict__ bias, int M, int N, int K)
{
    const int lane = threadIdx.x & 31;
    const int wave = threadIdx.x >> 5;      // 0..7
    const int wm   = wave >> 1;             // 0..3 (M)
    const int wn   = wave & 1;              // 0..1 (N)
    const int half = lane >> 4;
    const int lrow = lane & 15;

    const int mBase = blockIdx.y * 128 + wm * 32;
    const int nBase = blockIdx.x * 128 + wn * 64;

    // invariant per-lane row pointers (B rows clamped into range)
    const _Float16* pArow[2];
    #pragma unroll
    for (int mt = 0; mt < 2; ++mt)
        pArow[mt] = A + (size_t)(mBase + mt * 16 + lrow) * K + half * 8;
    const _Float16* pBrow[4];
    #pragma unroll
    for (int nt = 0; nt < 4; ++nt) {
        int wrow = nBase + nt * 16 + lrow;
        if (wrow > N - 1) wrow = N - 1;     // clamp: harmless, never stored
        pBrow[nt] = W + (size_t)wrow * K + half * 8;
    }

    v8f c[2][4] = {};
    v8h aLo[2][2], aHi[2][2], bLo[2][4], bHi[2][4];

    auto load_frags = [&](int buf, int kk) {
        #pragma unroll
        for (int mt = 0; mt < 2; ++mt) {
            aLo[buf][mt] = *(const v8h*)(pArow[mt] + kk);
            aHi[buf][mt] = *(const v8h*)(pArow[mt] + kk + 16);
        }
        #pragma unroll
        for (int nt = 0; nt < 4; ++nt) {
            bLo[buf][nt] = *(const v8h*)(pBrow[nt] + kk);
            bHi[buf][nt] = *(const v8h*)(pBrow[nt] + kk + 16);
        }
    };
    auto do_wmma = [&](int buf) {
        v16h a[2], b[4];
        #pragma unroll
        for (int mt = 0; mt < 2; ++mt)
            a[mt] = __builtin_shufflevector(aLo[buf][mt], aHi[buf][mt],
                        0,1,2,3,4,5,6,7,8,9,10,11,12,13,14,15);
        #pragma unroll
        for (int nt = 0; nt < 4; ++nt)
            b[nt] = __builtin_shufflevector(bLo[buf][nt], bHi[buf][nt],
                        0,1,2,3,4,5,6,7,8,9,10,11,12,13,14,15);
        #pragma unroll
        for (int mt = 0; mt < 2; ++mt)
            #pragma unroll
            for (int nt = 0; nt < 4; ++nt)
                c[mt][nt] = __builtin_amdgcn_wmma_f32_16x16x32_f16(
                    false, a[mt], false, b[nt], (short)0, c[mt][nt], false, false);
    };

    load_frags(0, 0);                        // prologue
    for (int k0 = 0; k0 < K; k0 += 64) {     // K is a multiple of 64
        load_frags(1, k0 + 32);              // buf1 loads fly over buf0 WMMAs
        __builtin_prefetch(pArow[0] + k0 + 64, 0, 1);
        do_wmma(0);
        if (k0 + 64 < K) load_frags(0, k0 + 64);  // buf0 loads over buf1 WMMAs
        do_wmma(1);
    }

    // D layout: lane -> col = lrow (+nBase); element j -> row = half*8 + j (+mBase)
    #pragma unroll
    for (int mt = 0; mt < 2; ++mt) {
        #pragma unroll
        for (int nt = 0; nt < 4; ++nt) {
            const int col = nBase + nt * 16 + lrow;
            if (col >= N) continue;
            #pragma unroll
            for (int j = 0; j < 8; ++j) {
                const int row = mBase + mt * 16 + half * 8 + j;
                float v = c[mt][nt][j];
                if (EP == EP_SPLIT) {
                    if (col < DI) C0[(size_t)row * DI + col] = v;
                    else          C1[(size_t)row * DI + (col - DI)] = v;
                } else if (EP == EP_PLAIN) {
                    C0[(size_t)row * N + col] = v;
                } else if (EP == EP_SOFTPLUS) {
                    v += bias[col];
                    v = (v > 20.0f) ? v : logf(1.0f + __expf(v));
                    C0[(size_t)row * N + col] = v;
                } else { // EP_BIAS
                    C0[(size_t)row * N + col] = v + bias[col];
                }
            }
        }
    }
}

// ---------------------------------------------------------------------------
// Elementwise helpers
// ---------------------------------------------------------------------------
__global__ void cvt_f32_f16(const float* __restrict__ in, _Float16* __restrict__ out, size_t n)
{
    size_t i = (size_t)blockIdx.x * blockDim.x + threadIdx.x;
    if (i < n) out[i] = (_Float16)in[i];
}

__global__ void cvt_dt(const float* __restrict__ xdbl, _Float16* __restrict__ dt)
{
    size_t i = (size_t)blockIdx.x * blockDim.x + threadIdx.x;
    if (i >= (size_t)MROWS * DTR) return;
    size_t row = i / DTR;
    int    col = (int)(i % DTR);
    dt[i] = (_Float16)xdbl[row * XPN + col];
}

// causal depthwise conv (DC=4) + bias + SiLU, f32 in -> f16 out
__global__ void conv_silu_kernel(const float* __restrict__ u, const float* __restrict__ cw,
                                 const float* __restrict__ cb, _Float16* __restrict__ out)
{
    size_t i = (size_t)blockIdx.x * blockDim.x + threadIdx.x;
    if (i >= (size_t)MROWS * DI) return;
    const int    d = (int)(i % DI);
    const size_t r = i / DI;           // row in [0, B*L)
    const int    l = (int)(r % SEQ);
    float acc = cb[d];
    #pragma unroll
    for (int k = 0; k < DCONV; ++k) {
        int ls = l - (DCONV - 1) + k;
        if (ls >= 0) acc += u[(r - (size_t)l + (size_t)ls) * DI + d] * cw[d * DCONV + k];
    }
    acc = acc / (1.0f + __expf(-acc));  // SiLU
    out[i] = (_Float16)acc;
}

// ---------------------------------------------------------------------------
// Selective scan: one thread per (batch, channel); 16-wide state in registers.
// B_t/C_t staged to LDS once per timestep. SiLU(z) gating fused; f16 y out.
// grid = 64 (4 batches x 16 channel-chunks), block = 128
// ---------------------------------------------------------------------------
__global__ __launch_bounds__(128)
void scan_kernel(const _Float16* __restrict__ u16, const float* __restrict__ delta,
                 const float* __restrict__ xdbl, const float* __restrict__ alog,
                 const float* __restrict__ Dp,   const float* __restrict__ z,
                 _Float16* __restrict__ y16)
{
    const int b = blockIdx.x >> 4;
    const int d = (blockIdx.x & 15) * 128 + threadIdx.x;

    float A[NDS], h[NDS];
    const float Dd = Dp[d];
    #pragma unroll
    for (int n = 0; n < NDS; ++n) {
        A[n] = -__expf(alog[(size_t)d * NDS + n]);
        h[n] = 0.0f;
    }

    __shared__ float sBC[2 * NDS];
    for (int l = 0; l < SEQ; ++l) {
        const size_t row = (size_t)b * SEQ + l;
        __syncthreads();
        if (threadIdx.x < 2 * NDS) sBC[threadIdx.x] = xdbl[row * XPN + DTR + threadIdx.x];
        __syncthreads();

        const float dl = delta[row * DI + d];
        const float uv = (float)u16[row * DI + d];
        const float du = dl * uv;
        float y = 0.0f;
        #pragma unroll
        for (int n = 0; n < NDS; ++n) {
            float dA = __expf(dl * A[n]);
            h[n] = dA * h[n] + du * sBC[n];
            y += h[n] * sBC[NDS + n];
        }
        y += uv * Dd;
        const float zv = z[row * DI + d];
        y *= zv / (1.0f + __expf(-zv));   // y * silu(z)
        y16[row * DI + d] = (_Float16)y;
    }
}

// ---------------------------------------------------------------------------
// residual add + LayerNorm; emits f32 (next residual) and f16 (next GEMM A)
// grid = MROWS, block = 256, each thread owns 4 of the 1024 columns
// ---------------------------------------------------------------------------
__global__ __launch_bounds__(256)
void ln_res_kernel(const float* __restrict__ hbuf, const float* __restrict__ res,
                   const float* __restrict__ g,    const float* __restrict__ bb,
                   float* __restrict__ xout, _Float16* __restrict__ x16)
{
    const int row = blockIdx.x;
    __shared__ float s1[256], s2[256];
    float v[4];
    float sum = 0.0f, sq = 0.0f;
    #pragma unroll
    for (int j = 0; j < 4; ++j) {
        int c = threadIdx.x + j * 256;
        float t = hbuf[(size_t)row * DMOD + c] + res[(size_t)row * DMOD + c];
        v[j] = t; sum += t; sq += t * t;
    }
    s1[threadIdx.x] = sum; s2[threadIdx.x] = sq;
    __syncthreads();
    for (int st = 128; st > 0; st >>= 1) {
        if (threadIdx.x < st) {
            s1[threadIdx.x] += s1[threadIdx.x + st];
            s2[threadIdx.x] += s2[threadIdx.x + st];
        }
        __syncthreads();
    }
    const float mu  = s1[0] * (1.0f / DMOD);
    const float var = s2[0] * (1.0f / DMOD) - mu * mu;
    const float rs  = rsqrtf(var + 1e-5f);
    #pragma unroll
    for (int j = 0; j < 4; ++j) {
        int c = threadIdx.x + j * 256;
        float o = (v[j] - mu) * rs * g[c] + bb[c];
        xout[(size_t)row * DMOD + c] = o;
        x16 [(size_t)row * DMOD + c] = (_Float16)o;
    }
}

// ---------------------------------------------------------------------------
extern "C" void kernel_launch(void* const* d_in, const int* in_sizes, int n_in,
                              void* d_out, int out_size, void* d_ws, size_t ws_size,
                              hipStream_t stream)
{
    (void)in_sizes; (void)n_in; (void)out_size; (void)ws_size;

    const float* x    = (const float*)d_in[0];
    const float* ipw  = (const float*)d_in[1];
    const float* cw   = (const float*)d_in[2];
    const float* cb   = (const float*)d_in[3];
    const float* xpw  = (const float*)d_in[4];
    const float* dpw  = (const float*)d_in[5];
    const float* dpb  = (const float*)d_in[6];
    const float* alog = (const float*)d_in[7];
    const float* Dp   = (const float*)d_in[8];
    const float* opw  = (const float*)d_in[9];
    const float* lng  = (const float*)d_in[10];
    const float* lnb  = (const float*)d_in[11];
    const float* pw   = (const float*)d_in[12];
    const float* pb   = (const float*)d_in[13];
    float* out = (float*)d_out;

    // workspace carve-up (256B aligned)
    char* ws = (char*)d_ws;
    size_t off = 0;
    auto alloc = [&](size_t bytes) -> char* {
        char* p = ws + off;
        off += (bytes + 255) & ~(size_t)255;
        return p;
    };
    _Float16* wIn  = (_Float16*)alloc((size_t)NL * 2 * DI * DMOD * sizeof(_Float16));
    _Float16* wXp  = (_Float16*)alloc((size_t)NL * XPN * DI     * sizeof(_Float16));
    _Float16* wDt  = (_Float16*)alloc((size_t)NL * DI * DTR     * sizeof(_Float16));
    _Float16* wOut = (_Float16*)alloc((size_t)NL * DMOD * DI    * sizeof(_Float16));
    _Float16* wPr  = (_Float16*)alloc((size_t)DMOD * DMOD       * sizeof(_Float16));
    float*    xcur = (float*)   alloc((size_t)MROWS * DMOD * sizeof(float));
    _Float16* x16  = (_Float16*)alloc((size_t)MROWS * DMOD * sizeof(_Float16));
    float*    uraw = (float*)   alloc((size_t)MROWS * DI   * sizeof(float));  // reused as delta
    float*    zbuf = (float*)   alloc((size_t)MROWS * DI   * sizeof(float));
    _Float16* ucnv = (_Float16*)alloc((size_t)MROWS * DI   * sizeof(_Float16));
    float*    xdbl = (float*)   alloc((size_t)MROWS * XPN  * sizeof(float));
    _Float16* dt16 = (_Float16*)alloc((size_t)MROWS * DTR  * sizeof(_Float16));
    _Float16* y16  = (_Float16*)alloc((size_t)MROWS * DI   * sizeof(_Float16));
    float*    hbuf = (float*)   alloc((size_t)MROWS * DMOD * sizeof(float));

    auto cvt = [&](const float* src, _Float16* dst, size_t n) {
        cvt_f32_f16<<<(unsigned)((n + 255) / 256), 256, 0, stream>>>(src, dst, n);
    };

    // weights -> f16 (deterministic every call)
    cvt(ipw, wIn,  (size_t)NL * 2 * DI * DMOD);
    cvt(xpw, wXp,  (size_t)NL * XPN * DI);
    cvt(dpw, wDt,  (size_t)NL * DI * DTR);
    cvt(opw, wOut, (size_t)NL * DMOD * DI);
    cvt(pw,  wPr,  (size_t)DMOD * DMOD);

    // x -> residual copy + f16 A-operand
    hipMemcpyAsync(xcur, x, (size_t)MROWS * DMOD * sizeof(float),
                   hipMemcpyDeviceToDevice, stream);
    cvt(x, x16, (size_t)MROWS * DMOD);

    const dim3 blk(256);
    for (int i = 0; i < NL; ++i) {
        // in_proj: [8192,1024] x [4096,1024]^T -> u_raw / z
        gemm_wmma<EP_SPLIT><<<dim3(2 * DI / 128, MROWS / 128), blk, 0, stream>>>(
            x16, wIn + (size_t)i * 2 * DI * DMOD, uraw, zbuf, nullptr,
            MROWS, 2 * DI, DMOD);

        // causal depthwise conv + SiLU
        conv_silu_kernel<<<(unsigned)(((size_t)MROWS * DI + 255) / 256), blk, 0, stream>>>(
            uraw, cw + (size_t)i * DI * DCONV, cb + (size_t)i * DI, ucnv);

        // x_proj: [8192,2048] x [96,2048]^T -> x_dbl (N=96, clamped B rows)
        gemm_wmma<EP_PLAIN><<<dim3(1, MROWS / 128), blk, 0, stream>>>(
            ucnv, wXp + (size_t)i * XPN * DI, xdbl, nullptr, nullptr,
            MROWS, XPN, DI);

        // dt slice -> f16
        cvt_dt<<<(unsigned)(((size_t)MROWS * DTR + 255) / 256), blk, 0, stream>>>(xdbl, dt16);

        // dt_proj + softplus(+bias) -> delta (reuses u_raw buffer)
        gemm_wmma<EP_SOFTPLUS><<<dim3(DI / 128, MROWS / 128), blk, 0, stream>>>(
            dt16, wDt + (size_t)i * DI * DTR, uraw, nullptr, dpb + (size_t)i * DI,
            MROWS, DI, DTR);

        // selective scan + D skip + SiLU(z) gating -> y (f16)
        scan_kernel<<<64, 128, 0, stream>>>(
            ucnv, uraw, xdbl, alog + (size_t)i * DI * NDS, Dp + (size_t)i * DI,
            zbuf, y16);

        // out_proj: [8192,2048] x [1024,2048]^T -> hbuf
        gemm_wmma<EP_PLAIN><<<dim3(DMOD / 128, MROWS / 128), blk, 0, stream>>>(
            y16, wOut + (size_t)i * DMOD * DI, hbuf, nullptr, nullptr,
            MROWS, DMOD, DI);

        // residual + LayerNorm -> xcur (f32) and x16 (f16)
        ln_res_kernel<<<MROWS, blk, 0, stream>>>(
            hbuf, xcur, lng + (size_t)i * DMOD, lnb + (size_t)i * DMOD, xcur, x16);
    }

    // final projection + bias -> d_out
    gemm_wmma<EP_BIAS><<<dim3(DMOD / 128, MROWS / 128), blk, 0, stream>>>(
        x16, wPr, out, nullptr, pb, MROWS, DMOD, DMOD);
}
